// GPT2AttentionBlock_51470888075453
// MI455X (gfx1250) — matrix operand order repaired and checked by reading, not compiled, not run on hardware
//
#include <hip/hip_runtime.h>
#include <hip/hip_bf16.h>

#define B_   8
#define S_   1024
#define D_   768
#define H_   12
#define DH_  64
#define N3D  (3 * D_)
#define MROWS (B_ * S_)   // 8192

typedef __attribute__((ext_vector_type(16))) __bf16 v16bf;
typedef __attribute__((ext_vector_type(8)))  float  v8f;
typedef __attribute__((ext_vector_type(4)))  unsigned v4u;
typedef __attribute__((ext_vector_type(8)))  int      v8i;
typedef __attribute__((ext_vector_type(4)))  int      v4i;

union V16 { v16bf v; unsigned u[8]; };

__device__ __forceinline__ unsigned short f2bf(float f) {
    union { float f; unsigned u; } c; c.f = f;
    unsigned u = c.u;
    unsigned r = u + 0x7FFFu + ((u >> 16) & 1u);   // round-to-nearest-even
    return (unsigned short)(r >> 16);
}

// ---- Tensor Data Mover: 2-D tile (bf16) global -> LDS, D# packed per ISA §8 ----
__device__ __forceinline__ void tdm_load_2d_bf16(
    unsigned lds_off, unsigned long long gaddr,
    unsigned tensor_d0, unsigned tensor_d1, unsigned stride_d0,
    unsigned tile_d0, unsigned tile_d1)
{
    v4u g0;
    g0[0] = 1u;                                             // count=1 (valid), user mode
    g0[1] = lds_off;                                        // lds_addr (bytes)
    g0[2] = (unsigned)(gaddr & 0xFFFFFFFFull);              // global_addr[31:0]
    g0[3] = (unsigned)((gaddr >> 32) & 0x01FFFFFFull)       // global_addr[56:32]
            | (2u << 30);                                   // type=2 ("image")
    v8i g1;
    g1[0] = (int)(1u << 16);                                // data_size=1 -> 2 bytes
    g1[1] = (int)((tensor_d0 & 0xFFFFu) << 16);             // tensor_dim0[15:0]
    g1[2] = (int)(((tensor_d0 >> 16) & 0xFFFFu)             // tensor_dim0[31:16]
            | ((tensor_d1 & 0xFFFFu) << 16));               // tensor_dim1[15:0]
    g1[3] = (int)(((tensor_d1 >> 16) & 0xFFFFu)             // tensor_dim1[31:16]
            | ((tile_d0 & 0xFFFFu) << 16));                 // tile_dim0
    g1[4] = (int)(tile_d1 & 0xFFFFu);                       // tile_dim1 (tile_dim2=0)
    g1[5] = (int)stride_d0;                                 // tensor_dim0_stride[31:0]
    g1[6] = 0;                                              // stride hi / dim1_stride lo
    g1[7] = 0;
    v4i z4 = {0, 0, 0, 0};
    v8i z8 = {0, 0, 0, 0, 0, 0, 0, 0};
    __builtin_amdgcn_tensor_load_to_lds(g0, g1, z4, z4, z8, 0);
}

// ---------------- conversion kernels ----------------

__global__ void k_f32_to_bf16(const float* __restrict__ in,
                              unsigned short* __restrict__ out, int n) {
    int i = blockIdx.x * blockDim.x + threadIdx.x;
    if (i < n) out[i] = f2bf(in[i]);
}

// W is [K,N] row-major fp32; Wt is [N,K] row-major bf16 (transposed)
__global__ void k_transpose_to_bf16(const float* __restrict__ W,
                                    unsigned short* __restrict__ Wt,
                                    int K, int N) {
    int i = blockIdx.x * blockDim.x + threadIdx.x;
    if (i < N * K) {
        int n = i / K, k = i % K;
        Wt[i] = f2bf(W[(size_t)k * N + n]);
    }
}

// ---------------- GEMM: C[M,N] = A[M,K](bf16) * Bt[N,K](bf16)^T + bias ----------------
// Block = 8 waves = 128(M) x 64(N).  B tile (64 x 32k) staged in LDS by the TDM,
// double-buffered; A tiles register-resident and software-pipelined.
// grid: x = N/64, y = M/128.
// mode 0: QKV epilogue (q bf16 ws, k/v fp32 out + bf16 ws); mode 1: fp32 out + bias.

__global__ __launch_bounds__(256) void k_gemm_bf16(
    const unsigned short* __restrict__ A,    // [M,K]
    const unsigned short* __restrict__ Bt,   // [N,K]
    const float* __restrict__ bias,          // [N]
    int M, int N, int K, int mode,
    float* __restrict__ outF,                // mode 1: [M,N]
    unsigned short* __restrict__ q_bf,       // mode 0: [M, D]
    float* __restrict__ k_out,               // mode 0: [B,H,S,DH]
    unsigned short* __restrict__ k_bf,       // mode 0: [B,H,S,DH]
    float* __restrict__ v_out,               // mode 0: [B,H,S,DH]
    unsigned short* __restrict__ vT_bf)      // mode 0: [B,H,DH,S]
{
    __shared__ unsigned short bbuf[2][64 * 32];   // only shared object -> LDS offset 0

    int wid    = threadIdx.x >> 5;
    int lane   = threadIdx.x & 31;
    int laneHi = lane >> 4;
    int lan    = lane & 15;

    int n0 = blockIdx.x * 64;
    int m0 = blockIdx.y * 128;

    const unsigned short* Arow = A + (size_t)(m0 + wid * 16 + lan) * K;

    v8f c[4];
#pragma unroll
    for (int j = 0; j < 4; ++j) c[j] = (v8f){0.f,0.f,0.f,0.f,0.f,0.f,0.f,0.f};

    int nK = K >> 5;   // K / 32 steps

    // prologue: TDM buffer 0, prefetch first A tile
    if (wid == 0) {
        tdm_load_2d_bf16(0u,
            (unsigned long long)(const void*)Bt + ((size_t)n0 * K) * 2ull,
            (unsigned)K, (unsigned)N, (unsigned)K, 32u, 64u);
    }
    V16 acur, anxt;
#pragma unroll
    for (int p = 0; p < 8; ++p) {
        int kp = 2 * p + laneHi * 16;
        acur.u[p] = *reinterpret_cast<const unsigned*>(Arow + kp);
    }

    for (int kt = 0; kt < nK; ++kt) {
        if (wid == 0) __builtin_amdgcn_s_wait_tensorcnt(0);
        __syncthreads();   // publish buf[kt&1]; all readers of buf[(kt+1)&1] are done
        if (wid == 0 && kt + 1 < nK) {
            tdm_load_2d_bf16((unsigned)(((kt + 1) & 1) * 64 * 32 * 2),
                (unsigned long long)(const void*)Bt +
                    ((size_t)n0 * K + (size_t)(kt + 1) * 32) * 2ull,
                (unsigned)K, (unsigned)N, (unsigned)K, 32u, 64u);
        }
        if (kt + 1 < nK) {     // software-pipelined A prefetch
            int kb = (kt + 1) * 32;
#pragma unroll
            for (int p = 0; p < 8; ++p) {
                int kp = 2 * p + laneHi * 16 + kb;
                anxt.u[p] = *reinterpret_cast<const unsigned*>(Arow + kp);
            }
        }
        const unsigned short* bb = &bbuf[kt & 1][0];
        // TDM (not the shader) wrote bbuf; tell the compiler LDS was modified so
        // it cannot fold the ds_loads to undef or dedupe the four B tiles.
        asm volatile("" : : "r"(bb) : "memory");
        // stage all four B tiles first -> single dscnt wait, then 4 back-to-back WMMAs
        V16 bj[4];
#pragma unroll
        for (int j = 0; j < 4; ++j) {
            const unsigned* bp = reinterpret_cast<const unsigned*>(
                bb + ((j * 16 + lan) * 32 + laneHi * 16));
#pragma unroll
            for (int g = 0; g < 8; ++g) bj[j].u[g] = bp[g];
        }
#pragma unroll
        for (int j = 0; j < 4; ++j)
            c[j] = __builtin_amdgcn_wmma_f32_16x16x32_bf16(
                       false, acur.v, false, bj[j].v, (short)0, c[j], false, false);
        acur = anxt;
    }

#pragma unroll
    for (int j = 0; j < 4; ++j) {
        int n = n0 + j * 16 + lan;
        float bv = bias[n];
#pragma unroll
        for (int r = 0; r < 8; ++r) {
            int m = m0 + wid * 16 + r + laneHi * 8;
            float val = c[j][r] + bv;
            if (mode == 1) {
                outF[(size_t)m * N + n] = val;
            } else {
                int b = m >> 10, s = m & (S_ - 1);
                if (n < D_) {
                    q_bf[(size_t)m * D_ + n] = f2bf(val);
                } else if (n < 2 * D_) {
                    int nk = n - D_; int h = nk >> 6, dh = nk & 63;
                    size_t idx = ((size_t)(b * H_ + h) * S_ + s) * DH_ + dh;
                    k_out[idx] = val;
                    k_bf[idx]  = f2bf(val);
                } else {
                    int nv = n - 2 * D_; int h = nv >> 6, dh = nv & 63;
                    v_out[((size_t)(b * H_ + h) * S_ + s) * DH_ + dh] = val;
                    vT_bf[((size_t)(b * H_ + h) * DH_ + dh) * S_ + s] = f2bf(val);
                }
            }
        }
    }
}

// ---------------- causal flash attention ----------------
// One wave per (b, h, 16-query tile). Keys processed in blocks of 32.

__global__ __launch_bounds__(256) void k_attn(
    const unsigned short* __restrict__ q_bf,   // [B*S, D]
    const unsigned short* __restrict__ k_bf,   // [B,H,S,DH]
    const unsigned short* __restrict__ vT_bf,  // [B,H,DH,S]
    unsigned short* __restrict__ attn_bf)      // [B*S, D]
{
    __shared__ unsigned short pbuf[8][16 * 32];

    int wid    = threadIdx.x >> 5;
    int lane   = threadIdx.x & 31;
    int laneHi = lane >> 4;
    int lan    = lane & 15;

    int gw = blockIdx.x * 8 + wid;
    int qt = gw & 63;
    int bh = gw >> 6;                 // b*H + h
    int b  = bh / H_, h = bh % H_;
    int qbase = qt * 16;

    // Q tile as two 16x32 A matrices (dh 0..31 and 32..63)
    V16 a0, a1;
    const unsigned short* qrow =
        q_bf + (size_t)(b * S_ + qbase + lan) * D_ + h * DH_;
#pragma unroll
    for (int p = 0; p < 8; ++p) {
        int kp = 2 * p + laneHi * 16;
        a0.u[p] = *reinterpret_cast<const unsigned*>(qrow + kp);
        a1.u[p] = *reinterpret_cast<const unsigned*>(qrow + kp + 32);
    }

    float mrun[8], lrun[8];
#pragma unroll
    for (int r = 0; r < 8; ++r) { mrun[r] = -3.0e38f; lrun[r] = 0.f; }
    v8f o[4];
#pragma unroll
    for (int j = 0; j < 4; ++j) o[j] = (v8f){0.f,0.f,0.f,0.f,0.f,0.f,0.f,0.f};

    const unsigned short* kb_p = k_bf  + (size_t)bh * S_ * DH_;
    const unsigned short* vb_p = vT_bf + (size_t)bh * DH_ * S_;

    int nblk = (qbase + 47) >> 5;
    for (int blk = 0; blk < nblk; ++blk) {
        int kb = blk * 32;

        // scores: two 16x16 tiles (keys kb..kb+15, kb+16..kb+31)
        v8f s0 = (v8f){0.f,0.f,0.f,0.f,0.f,0.f,0.f,0.f};
        v8f s1 = (v8f){0.f,0.f,0.f,0.f,0.f,0.f,0.f,0.f};
#pragma unroll
        for (int t = 0; t < 2; ++t) {
            const unsigned short* kc =
                kb_p + (size_t)(kb + t * 16 + lan) * DH_ + laneHi * 16;
            V16 b0, b1;
#pragma unroll
            for (int g = 0; g < 8; ++g) {
                b0.u[g] = *reinterpret_cast<const unsigned*>(kc + 2 * g);
                b1.u[g] = *reinterpret_cast<const unsigned*>(kc + 32 + 2 * g);
            }
            v8f acc = (t == 0) ? s0 : s1;
            acc = __builtin_amdgcn_wmma_f32_16x16x32_bf16(
                      false, a0.v, false, b0.v, (short)0, acc, false, false);
            acc = __builtin_amdgcn_wmma_f32_16x16x32_bf16(
                      false, a1.v, false, b1.v, (short)0, acc, false, false);
            if (t == 0) s0 = acc; else s1 = acc;
        }

        int col0 = kb + lan, col1 = col0 + 16;
#pragma unroll
        for (int r = 0; r < 8; ++r) {
            int qg = qbase + r + laneHi * 8;
            float v0 = s0[r] * 0.125f; if (col0 > qg) v0 = -3.0e38f;
            float v1 = s1[r] * 0.125f; if (col1 > qg) v1 = -3.0e38f;
            float mx = fmaxf(v0, v1);
#pragma unroll
            for (int d = 1; d < 16; d <<= 1) mx = fmaxf(mx, __shfl_xor(mx, d, 32));
            float mnew  = fmaxf(mrun[r], mx);
            float alpha = __expf(mrun[r] - mnew);
            float p0 = __expf(v0 - mnew), p1 = __expf(v1 - mnew);
            float rs = p0 + p1;
#pragma unroll
            for (int d = 1; d < 16; d <<= 1) rs += __shfl_xor(rs, d, 32);
            lrun[r] = lrun[r] * alpha + rs;
            mrun[r] = mnew;
#pragma unroll
            for (int j = 0; j < 4; ++j) o[j][r] *= alpha;
            pbuf[wid][(r + laneHi * 8) * 32 + lan]      = f2bf(p0);
            pbuf[wid][(r + laneHi * 8) * 32 + 16 + lan] = f2bf(p1);
        }
        asm volatile("s_wait_dscnt 0" ::: "memory");

        // reload P in A-matrix layout
        V16 ap;
#pragma unroll
        for (int p = 0; p < 8; ++p) {
            int kp = 2 * p + laneHi * 16;
            ap.u[p] = *reinterpret_cast<const unsigned*>(&pbuf[wid][lan * 32 + kp]);
        }

        // O += P * V  (4 dh tiles of 16)
#pragma unroll
        for (int j = 0; j < 4; ++j) {
            const unsigned short* vc =
                vb_p + (size_t)(j * 16 + lan) * S_ + kb + laneHi * 16;
            V16 bv;
#pragma unroll
            for (int g = 0; g < 8; ++g)
                bv.u[g] = *reinterpret_cast<const unsigned*>(vc + 2 * g);
            o[j] = __builtin_amdgcn_wmma_f32_16x16x32_bf16(
                       false, ap.v, false, bv.v, (short)0, o[j], false, false);
        }
    }

    // normalize and store bf16 for the projection GEMM
#pragma unroll
    for (int r = 0; r < 8; ++r) {
        float inv = 1.0f / lrun[r];
        int m = b * S_ + qbase + r + laneHi * 8;
#pragma unroll
        for (int j = 0; j < 4; ++j)
            attn_bf[(size_t)m * D_ + h * DH_ + j * 16 + lan] = f2bf(o[j][r] * inv);
    }
}

// ---------------- host launch ----------------

extern "C" void kernel_launch(void* const* d_in, const int* in_sizes, int n_in,
                              void* d_out, int out_size, void* d_ws, size_t ws_size,
                              hipStream_t stream) {
    const float* x     = (const float*)d_in[0];
    const float* Wqkv  = (const float*)d_in[1];
    const float* bqkv  = (const float*)d_in[2];
    const float* Wproj = (const float*)d_in[3];
    const float* bproj = (const float*)d_in[4];

    float* out   = (float*)d_out;                         // [B,S,D]
    float* k_out = out + (size_t)MROWS * D_;              // [B,H,S,DH]
    float* v_out = k_out + (size_t)MROWS * D_;            // [B,H,S,DH]

    char* ws = (char*)d_ws;
    size_t off = 0;
    unsigned short* x_bf    = (unsigned short*)(ws + off); off += (size_t)MROWS * D_ * 2;
    unsigned short* Wqkv_t  = (unsigned short*)(ws + off); off += (size_t)N3D * D_ * 2;
    unsigned short* Wproj_t = (unsigned short*)(ws + off); off += (size_t)D_ * D_ * 2;
    unsigned short* q_bf    = (unsigned short*)(ws + off); off += (size_t)MROWS * D_ * 2;
    unsigned short* k_bf    = (unsigned short*)(ws + off); off += (size_t)MROWS * D_ * 2;
    unsigned short* vT_bf   = (unsigned short*)(ws + off); off += (size_t)MROWS * D_ * 2;
    unsigned short* attn_bf = (unsigned short*)(ws + off); off += (size_t)MROWS * D_ * 2;

    int n1 = MROWS * D_;
    k_f32_to_bf16<<<(n1 + 255) / 256, 256, 0, stream>>>(x, x_bf, n1);
    int n2 = N3D * D_;
    k_transpose_to_bf16<<<(n2 + 255) / 256, 256, 0, stream>>>(Wqkv, Wqkv_t, D_, N3D);
    int n3 = D_ * D_;
    k_transpose_to_bf16<<<(n3 + 255) / 256, 256, 0, stream>>>(Wproj, Wproj_t, D_, D_);

    {   // QKV GEMM: M=8192, N=2304, K=768
        dim3 grid(N3D / 64, MROWS / 128);
        k_gemm_bf16<<<grid, 256, 0, stream>>>(
            x_bf, Wqkv_t, bqkv, MROWS, N3D, D_, 0,
            nullptr, q_bf, k_out, k_bf, v_out, vT_bf);
    }
    {   // attention: 8*12*64 = 6144 waves
        int waves = B_ * H_ * (S_ / 16);
        k_attn<<<waves / 8, 256, 0, stream>>>(q_bf, k_bf, vT_bf, attn_bf);
    }
    {   // projection GEMM: M=8192, N=768, K=768
        dim3 grid(D_ / 64, MROWS / 128);
        k_gemm_bf16<<<grid, 256, 0, stream>>>(
            attn_bf, Wproj_t, bproj, MROWS, D_, D_, 1,
            out, nullptr, nullptr, nullptr, nullptr, nullptr);
    }
}
